// GATNet_45500883533912
// MI455X (gfx1250) — matrix operand order repaired and checked by reading, not compiled, not run on hardware
//
#include <hip/hip_runtime.h>
#include <hip/hip_bf16.h>
#include <stdint.h>

// ---------------- problem constants (from reference setup_inputs) ------------
#define NNODES   65536      // B*Nper = 8*8192
#define NEDGES   1048576
#define CIN      128
#define NHEADS   4
#define C1       64
#define HC1      256        // NHEADS*C1
#define C2       16
#define HC2      64         // NHEADS*C2

typedef __attribute__((ext_vector_type(16))) _Float16 v16h;
typedef __attribute__((ext_vector_type(8)))  _Float16 v8h;
typedef __attribute__((ext_vector_type(4)))  _Float16 v4h;
typedef __attribute__((ext_vector_type(8)))  float    v8f;

// ---------------- helpers ----------------------------------------------------
__device__ __forceinline__ unsigned fflip(float f) {
    unsigned u = __float_as_uint(f);
    return (u & 0x80000000u) ? ~u : (u | 0x80000000u);
}
__device__ __forceinline__ float funflip(unsigned v) {
    return (v & 0x80000000u) ? __uint_as_float(v ^ 0x80000000u)
                             : __uint_as_float(~v);
}

// ---------------- zero fill --------------------------------------------------
__global__ __launch_bounds__(256) void zero_f32(float* __restrict__ p, long long n) {
    long long i = (long long)blockIdx.x * blockDim.x + threadIdx.x;
    if (i < n) p[i] = 0.0f;
}

// ---------------- fp32 -> f16 row-major convert (A operand) ------------------
__global__ __launch_bounds__(256) void conv_f16(const float* __restrict__ X,
                                                _Float16* __restrict__ Xh,
                                                long long n4) {  // n/4 threads
    long long i = (long long)blockIdx.x * blockDim.x + threadIdx.x;
    if (i >= n4) return;
    float4 v = *(const float4*)(X + i * 4);
    v4h h; h[0] = (_Float16)v.x; h[1] = (_Float16)v.y;
           h[2] = (_Float16)v.z; h[3] = (_Float16)v.w;
    *(v4h*)(Xh + i * 4) = h;
}

// ---------------- pack W[K,M] fp32 into WMMA B-fragment f16 layout -----------
// Fragment order: [kstep s][col-tile t][lane l] -> 16 contiguous halves, where
// lane l holds B[k = s*32 + (l>>4)*16 + i][col = t*16 + (l&15)], i = 0..15.
__global__ __launch_bounds__(256) void pack_w(const float* __restrict__ W,
                                              _Float16* __restrict__ Wp,
                                              int K, int M) {
    const int total = (K / 32) * (M / 16) * 32;
    const int idx = blockIdx.x * blockDim.x + threadIdx.x;
    if (idx >= total) return;
    const int l  = idx & 31;
    const int t  = (idx >> 5) % (M / 16);
    const int s  = idx / (32 * (M / 16));
    const int kb = s * 32 + (l >> 4) * 16;
    const int col = t * 16 + (l & 15);
    _Float16* o = Wp + (size_t)idx * 16;
    #pragma unroll
    for (int i = 0; i < 16; ++i)
        o[i] = (_Float16)W[(size_t)(kb + i) * M + col];
}

// ---------------- fused dual GEMM: XL = Xh@Wl, XR = Xh@Wr (WMMA f16->f32) ----
// Xh: [N, K] f16 row-major. Wlp/Wrp: packed fragments. XL/XR: [N, M] fp32.
// Block = 256 threads = 8 waves; block tile = 128 rows x 64 cols.
// B fragments staged via LDS (shared by all 8 waves).
__global__ __launch_bounds__(256) void gemm_dual_wmma(
        const _Float16* __restrict__ Xh,
        const _Float16* __restrict__ Wlp, const _Float16* __restrict__ Wrp,
        float* __restrict__ XL, float* __restrict__ XR,
        int K, int M) {
    __shared__ uint4 sB[2][4][32][2];   // [mat][tile][lane][2x16B] = 8 KB

    const int lane  = threadIdx.x & 31;
    const int wave  = threadIdx.x >> 5;
    const int row0  = (blockIdx.y * 8 + wave) * 16;
    const int col0  = blockIdx.x * 64;
    const int hsel  = lane >> 4;
    const int lrow  = lane & 15;
    const int arow  = row0 + lrow;
    const int ntcols = M / 16;          // col tiles in whole matrix

    // staging role for this thread (one 32B fragment-lane per thread)
    const int smat = threadIdx.x >> 7;          // 0 -> Wl, 1 -> Wr
    const int stile = (threadIdx.x >> 5) & 3;
    const int slane = threadIdx.x & 31;
    const _Float16* wsrc = smat ? Wrp : Wlp;

    v8f accL[4] = {}, accR[4] = {};

    for (int k0 = 0, s = 0; k0 < K; k0 += 32, ++s) {
        // ---- stage packed B fragments for this k-step into LDS ----
        {
            const uint4* p = (const uint4*)(wsrc +
                ((size_t)(s * ntcols + (col0 >> 4) + stile) * 32 + slane) * 16);
            sB[smat][stile][slane][0] = p[0];
            sB[smat][stile][slane][1] = p[1];
        }
        __syncthreads();

        // ---- A fragment: two contiguous 16B f16 loads ----
        const _Float16* ap = Xh + (size_t)arow * K + k0 + hsel * 16;
        v8h a0 = *(const v8h*)ap;
        v8h a1 = *(const v8h*)(ap + 8);
        v16h af = __builtin_shufflevector(a0, a1,
                    0, 1, 2, 3, 4, 5, 6, 7, 8, 9, 10, 11, 12, 13, 14, 15);

        #pragma unroll
        for (int t = 0; t < 4; ++t) {
            const v8h* blp = (const v8h*)&sB[0][t][lane][0];
            const v8h* brp = (const v8h*)&sB[1][t][lane][0];
            v16h bl = __builtin_shufflevector(blp[0], blp[1],
                        0, 1, 2, 3, 4, 5, 6, 7, 8, 9, 10, 11, 12, 13, 14, 15);
            v16h br = __builtin_shufflevector(brp[0], brp[1],
                        0, 1, 2, 3, 4, 5, 6, 7, 8, 9, 10, 11, 12, 13, 14, 15);
            accL[t] = __builtin_amdgcn_wmma_f32_16x16x32_f16(
                false, af, false, bl, (short)0, accL[t], false, false);
            accR[t] = __builtin_amdgcn_wmma_f32_16x16x32_f16(
                false, af, false, br, (short)0, accR[t], false, false);
        }
        __syncthreads();
    }

    // C/D layout: VGPR r: lanes 0-15 -> row r, lanes 16-31 -> row r+8
    #pragma unroll
    for (int t = 0; t < 4; ++t) {
        const int col = col0 + t * 16 + lrow;
        #pragma unroll
        for (int r = 0; r < 8; ++r) {
            const size_t off = (size_t)(row0 + hsel * 8 + r) * M + col;
            XL[off] = accL[t][r];
            XR[off] = accR[t][r];
        }
    }
}

// ---------------- edge score + segment max (H=4 hardcoded) -------------------
__global__ __launch_bounds__(256) void edge_score_kernel(
        const float* __restrict__ xl, const float* __restrict__ xr,
        const long long* __restrict__ src, const long long* __restrict__ dst,
        const float* __restrict__ att,
        float* __restrict__ score, unsigned* __restrict__ menc, int C) {
    const int idx = blockIdx.x * blockDim.x + threadIdx.x;   // E*4 threads
    const int e = idx >> 2, h = idx & 3;
    const long long s = src[e], d = dst[e];
    const int HC = 4 * C;
    const float4* a = (const float4*)(xl + (size_t)s * HC + h * C);
    const float4* b = (const float4*)(xr + (size_t)d * HC + h * C);
    const float4* w = (const float4*)(att + h * C);
    float acc = 0.0f;
    for (int c = 0; c < (C >> 2); ++c) {
        float4 av = a[c], bv = b[c], wv = w[c];
        float vx = av.x + bv.x, vy = av.y + bv.y, vz = av.z + bv.z, vw = av.w + bv.w;
        vx = vx > 0.0f ? vx : 0.2f * vx;
        vy = vy > 0.0f ? vy : 0.2f * vy;
        vz = vz > 0.0f ? vz : 0.2f * vz;
        vw = vw > 0.0f ? vw : 0.2f * vw;
        acc += wv.x * vx + wv.y * vy + wv.z * vz + wv.w * vw;
    }
    score[idx] = acc;
    atomicMax(menc + (size_t)d * 4 + h, fflip(acc));
}

// ---------------- alpha = exp(score - m[dst]); denom += alpha ----------------
__global__ __launch_bounds__(256) void edge_alpha_kernel(
        float* __restrict__ score, const unsigned* __restrict__ menc,
        float* __restrict__ denom, const long long* __restrict__ dst) {
    const int idx = blockIdx.x * blockDim.x + threadIdx.x;   // E*4 threads
    const int e = idx >> 2, h = idx & 3;
    const long long d = dst[e];
    const unsigned v = menc[(size_t)d * 4 + h];
    const float m = (v == 0u) ? 0.0f : funflip(v);   // no-edge node -> m = 0
    const float a = expf(score[idx] - m);
    score[idx] = a;                                  // reuse buffer as alpha
    atomicAdd(denom + (size_t)d * 4 + h, a);
}

// ---------------- scatter: accum[dst] += alpha * xl[src]  (float4/thread) ----
__global__ __launch_bounds__(256) void edge_aggregate_kernel(
        const float* __restrict__ alpha, const float* __restrict__ xl,
        const long long* __restrict__ src, const long long* __restrict__ dst,
        float* __restrict__ accum, int hcShift, int cShift) {
    const int idx = blockIdx.x * blockDim.x + threadIdx.x;   // E * HC/4 threads
    const int qShift = hcShift - 2;
    const int e  = idx >> qShift;
    const int rc = (idx & ((1 << qShift) - 1)) << 2;
    const long long s = src[e], d = dst[e];
    const float a = alpha[(e << 2) + (rc >> cShift)];
    float4 v = *(const float4*)(xl + ((size_t)s << hcShift) + rc);
    float* o = accum + ((size_t)d << hcShift) + rc;
    atomicAdd(o + 0, a * v.x);
    atomicAdd(o + 1, a * v.y);
    atomicAdd(o + 2, a * v.z);
    atomicAdd(o + 3, a * v.w);
}

// ---------------- node finalize: /denom + bias, ELU (in place) ---------------
__global__ __launch_bounds__(256) void node_finalize_kernel(
        float* __restrict__ accum, const float* __restrict__ denom,
        const float* __restrict__ bias, int hcShift, int cShift) {
    const int idx = blockIdx.x * blockDim.x + threadIdx.x;   // N * HC threads
    const int n  = idx >> hcShift;
    const int rc = idx & ((1 << hcShift) - 1);
    const int h  = rc >> cShift;
    const float dn = fmaxf(denom[(size_t)n * 4 + h], 1e-16f);
    float v = accum[idx] / dn + bias[rc];
    accum[idx] = v > 0.0f ? v : (expf(v) - 1.0f);            // ELU
}

// ---------------- launch -----------------------------------------------------
extern "C" void kernel_launch(void* const* d_in, const int* in_sizes, int n_in,
                              void* d_out, int out_size, void* d_ws, size_t ws_size,
                              hipStream_t stream) {
    (void)in_sizes; (void)n_in; (void)out_size; (void)ws_size;
    const float*     x    = (const float*)d_in[0];
    const long long* ei   = (const long long*)d_in[1];
    const float*     Wl1  = (const float*)d_in[2];
    const float*     Wr1  = (const float*)d_in[3];
    const float*     att1 = (const float*)d_in[4];
    const float*     b1   = (const float*)d_in[5];
    const float*     Wl2  = (const float*)d_in[6];
    const float*     Wr2  = (const float*)d_in[7];
    const float*     att2 = (const float*)d_in[8];
    const float*     b2   = (const float*)d_in[9];
    float* out = (float*)d_out;

    const long long* src = ei;
    const long long* dst = ei + NEDGES;

    // workspace layout (float = 4B units)
    float* ws    = (float*)d_ws;
    float* xl1   = ws;                              // N*HC1
    float* xr1   = xl1 + (size_t)NNODES * HC1;      // N*HC1
    float* h1    = xr1 + (size_t)NNODES * HC1;      // N*HC1 (accum -> features)
    float* sc    = h1  + (size_t)NNODES * HC1;      // E*4 score/alpha
    unsigned* me = (unsigned*)(sc + (size_t)NEDGES * 4);   // N*4
    float* dn    = (float*)(me + (size_t)NNODES * 4);      // N*4
    _Float16* Xh = (_Float16*)(dn + (size_t)NNODES * 4);   // N*HC1 halves (max)
    _Float16* Wlp = Xh + (size_t)NNODES * HC1;             // <= 32768 halves
    _Float16* Wrp = Wlp + 32768;
    float* xl2   = xl1;   // reuse (N*HC2 <= N*HC1)
    float* xr2   = xr1;

    const int B = 256;

    // ===== layer 1 =====
    {   long long n4 = (long long)NNODES * CIN / 4;
        conv_f16<<<(unsigned)((n4 + B - 1) / B), B, 0, stream>>>(x, Xh, n4); }
    pack_w<<<(unsigned)(((CIN / 32) * (HC1 / 16) * 32 + B - 1) / B), B, 0, stream>>>(Wl1, Wlp, CIN, HC1);
    pack_w<<<(unsigned)(((CIN / 32) * (HC1 / 16) * 32 + B - 1) / B), B, 0, stream>>>(Wr1, Wrp, CIN, HC1);
    gemm_dual_wmma<<<dim3(HC1 / 64, NNODES / 128), B, 0, stream>>>(
        Xh, Wlp, Wrp, xl1, xr1, CIN, HC1);
    {   // zero h1 + score + menc + denom (contiguous span)
        long long nz = (long long)NNODES * HC1 + (long long)NEDGES * 4
                     + (long long)NNODES * 4 + (long long)NNODES * 4;
        zero_f32<<<(unsigned)((nz + B - 1) / B), B, 0, stream>>>(h1, nz);
    }
    edge_score_kernel<<<NEDGES * 4 / B, B, 0, stream>>>(xl1, xr1, src, dst, att1, sc, me, C1);
    edge_alpha_kernel<<<NEDGES * 4 / B, B, 0, stream>>>(sc, me, dn, dst);
    edge_aggregate_kernel<<<(unsigned)((size_t)NEDGES * HC1 / 4 / B), B, 0, stream>>>(
        sc, xl1, src, dst, h1, /*hcShift=*/8, /*cShift=*/6);
    node_finalize_kernel<<<NNODES * HC1 / B, B, 0, stream>>>(h1, dn, b1, 8, 6);

    // ===== layer 2 =====
    {   long long n4 = (long long)NNODES * HC1 / 4;
        conv_f16<<<(unsigned)((n4 + B - 1) / B), B, 0, stream>>>(h1, Xh, n4); }
    pack_w<<<(unsigned)(((HC1 / 32) * (HC2 / 16) * 32 + B - 1) / B), B, 0, stream>>>(Wl2, Wlp, HC1, HC2);
    pack_w<<<(unsigned)(((HC1 / 32) * (HC2 / 16) * 32 + B - 1) / B), B, 0, stream>>>(Wr2, Wrp, HC1, HC2);
    gemm_dual_wmma<<<dim3(HC2 / 64, NNODES / 128), B, 0, stream>>>(
        Xh, Wlp, Wrp, xl2, xr2, HC1, HC2);
    {   // zero menc + denom
        long long nz = (long long)NNODES * 8;
        zero_f32<<<(unsigned)((nz + B - 1) / B), B, 0, stream>>>((float*)me, nz);
    }
    {   // zero output accumulator
        long long nz = (long long)NNODES * HC2;
        zero_f32<<<(unsigned)((nz + B - 1) / B), B, 0, stream>>>(out, nz);
    }
    edge_score_kernel<<<NEDGES * 4 / B, B, 0, stream>>>(xl2, xr2, src, dst, att2, sc, me, C2);
    edge_alpha_kernel<<<NEDGES * 4 / B, B, 0, stream>>>(sc, me, dn, dst);
    edge_aggregate_kernel<<<(unsigned)((size_t)NEDGES * HC2 / 4 / B), B, 0, stream>>>(
        sc, xl2, src, dst, out, /*hcShift=*/6, /*cShift=*/4);
    node_finalize_kernel<<<NNODES * HC2 / B, B, 0, stream>>>(out, dn, b2, 6, 4);
}
